// GGNNEncoder_23158463660091
// MI455X (gfx1250) — matrix operand order repaired and checked by reading, not compiled, not run on hardware
//
#include <hip/hip_runtime.h>
#include <hip/hip_bf16.h>

// GGNN encoder for MI455X (gfx1250): bf16 WMMA + async-to-LDS staging.
// B=4, N=1024, DIN=D=256, E=5, STEP=5.

typedef __bf16 bf16;
typedef __attribute__((ext_vector_type(16))) bf16 v16bf;
typedef __attribute__((ext_vector_type(8)))  float v8f;
typedef int i32x4 __attribute__((vector_size(16)));  // matches async-LDS builtin param

#define BDIM   4
#define NNODE  1024
#define DD     256
#define EE     5
#define NSTEP  5
#define KBIG   (EE * NNODE)   // 5120
#define NTILES (KBIG / 32)    // 160

// ---------- scalar helpers ----------
__device__ __forceinline__ bf16 f2bf(float x) {
  unsigned u = __float_as_uint(x);
  unsigned r = (u + 0x7FFFu + ((u >> 16) & 1u)) >> 16;  // RNE
  unsigned short s = (unsigned short)r;
  bf16 b;
  __builtin_memcpy(&b, &s, 2);
  return b;
}

__device__ __forceinline__ v8f wmma_bf16(v16bf a, v16bf b, v8f c) {
  return __builtin_amdgcn_wmma_f32_16x16x32_bf16(false, a, false, b, (short)0, c,
                                                 false, false);
}

// A fragment (16x32 bf16): lane = khalf*16 + row; element e -> K =
// khalf*16 + e. Two 16B loads.
__device__ __forceinline__ v16bf load_afrag(const bf16* p, int khalf) {
  union { v16bf v; uint4 q[2]; } u;
  u.q[0] = *(const uint4*)(p + khalf * 16);
  u.q[1] = *(const uint4*)(p + khalf * 16 + 8);
  return u.v;
}

// B fragment (32x16 bf16): col = lane&15, K base = (lane>>4)*16, K-contiguous.
__device__ __forceinline__ v16bf load_bfrag(const bf16* p) {
  union { v16bf v; uint4 q[2]; } u;
  u.q[0] = *(const uint4*)(p);
  u.q[1] = *(const uint4*)(p + 8);
  return u.v;
}

// ---------- async global -> LDS (gfx1250) ----------
__device__ __forceinline__ void async_b128(const bf16* g, bf16* l) {
#if __has_builtin(__builtin_amdgcn_global_load_async_to_lds_b128)
  __builtin_amdgcn_global_load_async_to_lds_b128(
      (i32x4*)(void*)g, (i32x4*)(void*)l, 0, 0);
#else
  unsigned loff = (unsigned)(size_t)l;  // low 32 bits of flat LDS addr = LDS offset
  asm volatile("global_load_async_to_lds_b128 %0, %1, off"
               :: "v"(loff), "v"(g) : "memory");
#endif
}

__device__ __forceinline__ void wait_async_le2() {
#if __has_builtin(__builtin_amdgcn_s_wait_asynccnt)
  __builtin_amdgcn_s_wait_asynccnt(2);
#else
  asm volatile("s_wait_asynccnt 0x2" ::: "memory");
#endif
}
__device__ __forceinline__ void wait_async_le0() {
#if __has_builtin(__builtin_amdgcn_s_wait_asynccnt)
  __builtin_amdgcn_s_wait_asynccnt(0);
#else
  asm volatile("s_wait_asynccnt 0x0" ::: "memory");
#endif
}

// ---------- 32x64-per-wave main loop, compile-time strides ----------
template <int LDA, int LDW, int KTOT>
__device__ __forceinline__ void mainloop32(const bf16* __restrict__ A,
                                           const bf16* __restrict__ W,
                                           int rowBase, int colBase, v8f acc[2][4]) {
  const int lane  = threadIdx.x & 31;
  const int rlane = lane & 15;
  const int khalf = lane >> 4;
  const bf16* a0 = A + (size_t)(rowBase + rlane) * LDA;
  const bf16* a1 = a0 + 16 * LDA;
  for (int kb = 0; kb < KTOT; kb += 32) {
    v16bf af0 = load_afrag(a0 + kb, khalf);
    v16bf af1 = load_afrag(a1 + kb, khalf);
#pragma unroll
    for (int j = 0; j < 4; ++j) {
      const bf16* wp = W + (size_t)(colBase + j * 16 + rlane) * LDW + kb + khalf * 16;
      v16bf bf = load_bfrag(wp);
      acc[0][j] = wmma_bf16(af0, bf, acc[0][j]);
      acc[1][j] = wmma_bf16(af1, bf, acc[1][j]);
    }
  }
}

__device__ __forceinline__ void zero_acc8(v8f acc[2][4]) {
#pragma unroll
  for (int g = 0; g < 2; ++g)
#pragma unroll
    for (int j = 0; j < 4; ++j)
#pragma unroll
      for (int r = 0; r < 8; ++r) acc[g][j][r] = 0.0f;
}

// ---------- generic GEMM: C = A * W^T (+bias); block = 128 rows x 64 cols ----------
template <int LDA, int LDW, int KTOT>
__global__ void __launch_bounds__(128) wmma_gemm_kernel(
    const bf16* __restrict__ A, const bf16* __restrict__ W,
    const float* __restrict__ bias,
    float* __restrict__ Cf, int ldc,
    bf16* __restrict__ Cb, int ldcb) {
  const int wave    = threadIdx.x >> 5;
  const int lane    = threadIdx.x & 31;
  const int rowBase = blockIdx.y * 128 + wave * 32;
  const int colBase = blockIdx.x * 64;

  v8f acc[2][4];
  zero_acc8(acc);
  mainloop32<LDA, LDW, KTOT>(A, W, rowBase, colBase, acc);

  const int rlane = lane & 15;
  const int half  = lane >> 4;
#pragma unroll
  for (int g = 0; g < 2; ++g)
#pragma unroll
    for (int j = 0; j < 4; ++j) {
      const int col = colBase + j * 16 + rlane;
      const float bv = bias ? bias[col] : 0.0f;
#pragma unroll
      for (int r = 0; r < 8; ++r) {
        const int m = rowBase + g * 16 + half * 8 + r;
        const float v = acc[g][j][r] + bv;
        if (Cf) Cf[(size_t)m * ldc + col] = v;
        if (Cb) Cb[(size_t)m * ldcb + col] = f2bf(v);
      }
    }
}

// ---------- he GEMM, transposed bf16 out: heT[b][d][e*N+m] ----------
__global__ void __launch_bounds__(128) wmma_heT_kernel(
    const bf16* __restrict__ hb, const bf16* __restrict__ ew, bf16* __restrict__ heT) {
  const int wave    = threadIdx.x >> 5;
  const int lane    = threadIdx.x & 31;
  const int rowBase = blockIdx.y * 128 + wave * 32;  // node m
  const int colBase = blockIdx.x * 64;               // channel d
  const int b = blockIdx.z / EE;
  const int e = blockIdx.z % EE;

  v8f acc[2][4];
  zero_acc8(acc);
  mainloop32<DD, DD, DD>(hb + (size_t)b * NNODE * DD, ew + (size_t)e * DD * DD,
                         rowBase, colBase, acc);

  const int rlane = lane & 15;
  const int half  = lane >> 4;
  const size_t baseB = (size_t)b * DD * KBIG + (size_t)e * NNODE;
#pragma unroll
  for (int g = 0; g < 2; ++g)
#pragma unroll
    for (int j = 0; j < 4; ++j) {
      const int d  = colBase + j * 16 + rlane;
      const int m0 = rowBase + g * 16 + half * 8;  // 8 consecutive rows
      union { uint4 q; bf16 h[8]; } pk;
#pragma unroll
      for (int r = 0; r < 8; ++r) pk.h[r] = f2bf(acc[g][j][r]);
      *(uint4*)(heT + baseB + (size_t)d * KBIG + m0) = pk.q;  // 16B contiguous
    }
}

// ---------- big GEMM: acc[b] = adj[b]([N, E*N]) @ heT[b]^T + sum_edge_b ----------
// B tile (64 cols x 32 K) double-buffered in LDS via async global->LDS copies.
__global__ void __launch_bounds__(128) wmma_adj_kernel(
    const bf16* __restrict__ adj,   // [B][E][N][N] bf16
    const bf16* __restrict__ heT,   // [B][D][E*N] bf16
    const float* __restrict__ bias, // sum_edge_b [D]
    float* __restrict__ Cf, bf16* __restrict__ Cb) {
  __shared__ bf16 sB[2][64 * 32];  // 2 x 4KB

  const int tid   = threadIdx.x;
  const int wave  = tid >> 5;
  const int lane  = tid & 31;
  const int rlane = lane & 15;
  const int khalf = lane >> 4;
  const int b       = blockIdx.z;
  const int rowBase = blockIdx.y * 128 + wave * 32;
  const int colBase = blockIdx.x * 64;

  const bf16* Wb = heT + (size_t)b * DD * KBIG;
  const bf16* Ab = adj + (size_t)b * EE * NNODE * NNODE;

  // staging map: thread -> (col, 16-element k-half); 2 async b128 per wave/tile
  const int scol = tid >> 1;
  const int sko  = (tid & 1) * 16;
  auto issue = [&](int tile, int buf) {
    const int kb = tile * 32;
    const bf16* g = Wb + (size_t)(colBase + scol) * KBIG + kb + sko;
    bf16* l = &sB[buf][scol * 32 + sko];
    async_b128(g, l);
    async_b128(g + 8, l + 8);
  };

  issue(0, 0);
  issue(1, 1);

  v8f acc[2][4];
  zero_acc8(acc);

  const bf16* arow0 = Ab + (size_t)(rowBase + rlane) * NNODE;
  const bf16* arow1 = arow0 + (size_t)16 * NNODE;

  for (int i = 0; i < NTILES; ++i) {
    if (i < NTILES - 1) wait_async_le2();
    else                wait_async_le0();
    __syncthreads();  // all waves' tile-i data resident in LDS

    const int e   = i >> 5;         // segment (edge type): no division
    const int kin = (i & 31) << 5;  // k within segment
    const size_t aoff = (size_t)e * NNODE * NNODE + kin;
    v16bf af0 = load_afrag(arow0 + aoff, khalf);
    v16bf af1 = load_afrag(arow1 + aoff, khalf);

    const bf16* ls = &sB[i & 1][0];
#pragma unroll
    for (int j = 0; j < 4; ++j) {
      v16bf bf = load_bfrag(ls + (j * 16 + rlane) * 32 + khalf * 16);
      acc[0][j] = wmma_bf16(af0, bf, acc[0][j]);
      acc[1][j] = wmma_bf16(af1, bf, acc[1][j]);
    }
    __syncthreads();  // all waves done reading buf (i&1) before it is rewritten
    if (i + 2 < NTILES) issue(i + 2, i & 1);
  }

  const size_t cz = (size_t)b * NNODE * DD;
#pragma unroll
  for (int g = 0; g < 2; ++g)
#pragma unroll
    for (int j = 0; j < 4; ++j) {
      const int col = colBase + j * 16 + rlane;
      const float bv = bias[col];
#pragma unroll
      for (int r = 0; r < 8; ++r) {
        const int m = rowBase + g * 16 + (lane >> 4) * 8 + r;
        const float v = acc[g][j][r] + bv;
        Cf[cz + (size_t)m * DD + col] = v;
        Cb[cz + (size_t)m * DD + col] = f2bf(v);
      }
    }
}

// ---------- elementwise / reduction kernels ----------
__global__ void cvt_f32_bf16(const float* __restrict__ src, bf16* __restrict__ dst,
                             long long n) {
  long long i = (long long)blockIdx.x * blockDim.x + threadIdx.x;
  if (i < n) dst[i] = f2bf(src[i]);
}

__global__ void sum_edge_bias(const float* __restrict__ eb, float* __restrict__ out) {
  const int d = threadIdx.x;
  float s = 0.0f;
#pragma unroll
  for (int e = 0; e < EE; ++e) s += eb[e * DD + d];
  out[d] = s;
}

__global__ void __launch_bounds__(256) gru_kernel(
    const float* __restrict__ gi, const float* __restrict__ gh,
    float* __restrict__ h, bf16* __restrict__ hb) {
  const int row = blockIdx.x;
  const int d   = threadIdx.x;
  const size_t o3 = (size_t)row * (3 * DD) + d;
  const size_t oh = (size_t)row * DD + d;
  const float ir = gi[o3],          hr = gh[o3];
  const float iz = gi[o3 + DD],     hz = gh[o3 + DD];
  const float in = gi[o3 + 2 * DD], hn = gh[o3 + 2 * DD];
  const float r = 1.0f / (1.0f + __expf(-(ir + hr)));
  const float z = 1.0f / (1.0f + __expf(-(iz + hz)));
  const float n = tanhf(in + r * hn);
  const float hnew = (1.0f - z) * n + z * h[oh];
  h[oh]  = hnew;
  hb[oh] = f2bf(hnew);
}

__global__ void __launch_bounds__(256) attn_kernel(
    const float* __restrict__ al, const float* __restrict__ fl,
    const float* __restrict__ h,
    float* __restrict__ a, float* __restrict__ p, float* __restrict__ out) {
  __shared__ float s[256];
  const int row = blockIdx.x;
  const int d   = threadIdx.x;
  const size_t o = (size_t)row * DD + d;
  const float av = 1.0f / (1.0f + __expf(-al[o]));
  const float pv = av * tanhf(fl[o]);
  a[o] = av;
  p[o] = pv;
  out[1024 + o] = h[o];  // node_states
  s[d] = av;
  __syncthreads();
  for (int st = 128; st > 0; st >>= 1) {
    if (d < st) s[d] += s[d + st];
    __syncthreads();
  }
  if (d == 0) out[1024 + (size_t)BDIM * NNODE * DD + row] = s[0] * (1.0f / DD);
}

__global__ void __launch_bounds__(256) gemb_kernel(
    const float* __restrict__ p, float* __restrict__ out) {
  __shared__ float s[256];
  const int b = blockIdx.x >> 8;
  const int d = blockIdx.x & 255;
  const int t = threadIdx.x;
  float sum = 0.0f;
  for (int n = t; n < NNODE; n += 256)
    sum += p[((size_t)b * NNODE + n) * DD + d];
  s[t] = sum;
  __syncthreads();
  for (int st = 128; st > 0; st >>= 1) {
    if (t < st) s[t] += s[t + st];
    __syncthreads();
  }
  if (t == 0) out[b * DD + d] = s[0];
}

// ---------- launcher ----------
extern "C" void kernel_launch(void* const* d_in, const int* in_sizes, int n_in,
                              void* d_out, int out_size, void* d_ws, size_t ws_size,
                              hipStream_t stream) {
  (void)in_sizes; (void)n_in; (void)out_size; (void)ws_size;
  const float* X   = (const float*)d_in[0];
  const float* ADJ = (const float*)d_in[1];
  const float* FCW = (const float*)d_in[2];
  const float* FCB = (const float*)d_in[3];
  const float* EW  = (const float*)d_in[4];
  const float* EB  = (const float*)d_in[5];
  const float* WIH = (const float*)d_in[6];
  const float* WHH = (const float*)d_in[7];
  const float* BIH = (const float*)d_in[8];
  const float* BHH = (const float*)d_in[9];
  const float* C1W = (const float*)d_in[10];
  const float* C1B = (const float*)d_in[11];
  const float* C2W = (const float*)d_in[12];
  const float* C2B = (const float*)d_in[13];
  float* out = (float*)d_out;

  char* w = (char*)d_ws;
  auto alloc = [&](size_t bytes) -> char* {
    char* r = w;
    w += (bytes + 255) & ~(size_t)255;
    return r;
  };
  bf16*  adjb = (bf16*)alloc((size_t)BDIM * EE * NNODE * NNODE * 2);
  bf16*  xb   = (bf16*)alloc((size_t)BDIM * NNODE * DD * 2);
  bf16*  fcwb = (bf16*)alloc((size_t)DD * DD * 2);
  bf16*  ewb  = (bf16*)alloc((size_t)EE * DD * DD * 2);
  bf16*  wihb = (bf16*)alloc((size_t)3 * DD * DD * 2);
  bf16*  whhb = (bf16*)alloc((size_t)3 * DD * DD * 2);
  bf16*  c1wb = (bf16*)alloc((size_t)DD * DD * 2);
  bf16*  c2wb = (bf16*)alloc((size_t)DD * DD * 2);
  float* h    = (float*)alloc((size_t)BDIM * NNODE * DD * 4);
  bf16*  hb   = (bf16*)alloc((size_t)BDIM * NNODE * DD * 2);
  bf16*  heT  = (bf16*)alloc((size_t)BDIM * DD * KBIG * 2);
  float* acc  = (float*)alloc((size_t)BDIM * NNODE * DD * 4);
  bf16*  accb = (bf16*)alloc((size_t)BDIM * NNODE * DD * 2);
  float* gi   = (float*)alloc((size_t)BDIM * NNODE * 3 * DD * 4);
  float* gh   = (float*)alloc((size_t)BDIM * NNODE * 3 * DD * 4);
  float* al   = (float*)alloc((size_t)BDIM * NNODE * DD * 4);
  float* fl   = (float*)alloc((size_t)BDIM * NNODE * DD * 4);
  float* sumb = (float*)alloc(DD * 4);

  auto cvt = [&](const float* s, bf16* d, long long n) {
    cvt_f32_bf16<<<dim3((unsigned)((n + 255) / 256)), dim3(256), 0, stream>>>(s, d, n);
  };
  cvt(ADJ, adjb, (long long)BDIM * EE * NNODE * NNODE);
  cvt(X, xb, (long long)BDIM * NNODE * DD);
  cvt(FCW, fcwb, (long long)DD * DD);
  cvt(EW, ewb, (long long)EE * DD * DD);
  cvt(WIH, wihb, (long long)3 * DD * DD);
  cvt(WHH, whhb, (long long)3 * DD * DD);
  cvt(C1W, c1wb, (long long)DD * DD);
  cvt(C2W, c2wb, (long long)DD * DD);
  sum_edge_bias<<<1, 256, 0, stream>>>(EB, sumb);

  // h = X @ fc_in_w^T + b   [4096,256]x[256,256]
  wmma_gemm_kernel<DD, DD, DD><<<dim3(DD / 64, BDIM * NNODE / 128, 1), 128, 0, stream>>>(
      xb, fcwb, FCB, h, DD, hb, DD);

  for (int s = 0; s < NSTEP; ++s) {
    // he[b,e] = h[b] @ edge_w[e]^T  (stored transposed bf16)
    wmma_heT_kernel<<<dim3(DD / 64, NNODE / 128, BDIM * EE), 128, 0, stream>>>(
        hb, ewb, heT);
    // acc[b] = adj[b] @ he + sum_edge_b   (K = 5120, async-LDS staged B)
    wmma_adj_kernel<<<dim3(DD / 64, NNODE / 128, BDIM), 128, 0, stream>>>(
        adjb, heT, sumb, acc, accb);
    // GRU gate GEMMs [4096,256]x[256,768]
    wmma_gemm_kernel<DD, DD, DD>
        <<<dim3(3 * DD / 64, BDIM * NNODE / 128, 1), 128, 0, stream>>>(
            accb, wihb, BIH, gi, 3 * DD, (bf16*)nullptr, 0);
    wmma_gemm_kernel<DD, DD, DD>
        <<<dim3(3 * DD / 64, BDIM * NNODE / 128, 1), 128, 0, stream>>>(
            hb, whhb, BHH, gh, 3 * DD, (bf16*)nullptr, 0);
    gru_kernel<<<BDIM * NNODE, DD, 0, stream>>>(gi, gh, h, hb);
  }

  // attention head
  wmma_gemm_kernel<DD, DD, DD><<<dim3(DD / 64, BDIM * NNODE / 128, 1), 128, 0, stream>>>(
      hb, c1wb, C1B, al, DD, (bf16*)nullptr, 0);
  wmma_gemm_kernel<DD, DD, DD><<<dim3(DD / 64, BDIM * NNODE / 128, 1), 128, 0, stream>>>(
      hb, c2wb, C2B, fl, DD, (bf16*)nullptr, 0);
  attn_kernel<<<BDIM * NNODE, DD, 0, stream>>>(al, fl, h, al, fl, out);
  gemb_kernel<<<BDIM * DD, 256, 0, stream>>>(fl, out);
}
